// MoE_31507880084113
// MI455X (gfx1250) — matrix operand order repaired and checked by reading, not compiled, not run on hardware
//
#include <hip/hip_runtime.h>

typedef __bf16 bf16;
typedef bf16  v16bf __attribute__((ext_vector_type(16)));
typedef float v8f   __attribute__((ext_vector_type(8)));

#define BATCH 8
#define NEXP  4
#define FLATD (3*384*384)
#define NCLS  14

// ---------------- zero output ----------------
__global__ void k_zero(float* __restrict__ out, int n) {
  int i = blockIdx.x * blockDim.x + threadIdx.x;
  if (i < n) out[i] = 0.0f;
}

// ---------------- weight f32 -> bf16 with K padded to 32, zero-filled ----------------
__global__ void k_cvtw(const float* __restrict__ src, bf16* __restrict__ dst,
                       int Ktot, int Kp, int total /* Cout*Kp */) {
  int i = blockIdx.x * blockDim.x + threadIdx.x;
  if (i >= total) return;
  int n = i / Kp, k = i - n * Kp;
  dst[i] = (k < Ktot) ? (bf16)src[(size_t)n * Ktot + k] : (bf16)0.f;
}

// ---------------- gating: logits -> softmax -> top-1 gates ----------------
__global__ void k_gate(const float* __restrict__ x, const float* __restrict__ wg,
                       float* __restrict__ gates) {
  __shared__ float red[NEXP][256];
  const int b = blockIdx.x, t = threadIdx.x;
  float acc[NEXP] = {0.f, 0.f, 0.f, 0.f};
  const float* xb = x + (size_t)b * FLATD;
  for (int i = t; i < FLATD; i += 256) {
    float xv = xb[i];
    const float* wr = wg + (size_t)i * NEXP;
#pragma unroll
    for (int e = 0; e < NEXP; ++e) acc[e] += xv * wr[e];
  }
  for (int e = 0; e < NEXP; ++e) red[e][t] = acc[e];
  __syncthreads();
  for (int s = 128; s > 0; s >>= 1) {
    if (t < s)
      for (int e = 0; e < NEXP; ++e) red[e][t] += red[e][t + s];
    __syncthreads();
  }
  if (t == 0) {
    float l[NEXP], mx = red[0][0];
    for (int e = 0; e < NEXP; ++e) { l[e] = red[e][0]; mx = fmaxf(mx, l[e]); }
    float p[NEXP], s = 0.f;
    for (int e = 0; e < NEXP; ++e) { p[e] = __expf(l[e] - mx); s += p[e]; }
    int am = 0; float pm = p[0] / s;
    for (int e = 1; e < NEXP; ++e) { float pe = p[e] / s; if (pe > pm) { pm = pe; am = e; } }
    float gk = pm / (pm + 1e-6f);   // K=1: gk = top / (top + eps)
    for (int e = 0; e < NEXP; ++e) gates[b * NEXP + e] = (e == am) ? gk : 0.f;
  }
}

// ---------------- implicit-GEMM conv, bf16 WMMA, BM=128 BN=64 BK=32 ----------------
// 256 threads = 8 waves; wave w owns M rows [w*16, w*16+16), computes 4 WMMA tiles
// (16x64) per K-step. Weights pre-converted to bf16, rows padded to Kp (mult of 32).
// B tiles are fetched with double-buffered GLOBAL_LOAD_ASYNC_TO_LDS_B128 (ASYNCcnt).
template <int KS>
__global__ __launch_bounds__(256)
void k_conv_wmma(const float* __restrict__ in, const bf16* __restrict__ wgt,
                 const float* __restrict__ scale, const float* __restrict__ bias,
                 const float* __restrict__ resid, float* __restrict__ out,
                 int Cin, int IH, int IW, int Cout, int OH, int OW,
                 int stride, int pad, int relu, int Kp)
{
  constexpr int KK = KS * KS;
  const int Ktot = Cin * KK;
  const int HW = OH * OW;

  __shared__ bf16 As[128][40];      // [m][k], 80B row stride -> 16B-aligned chunks
  __shared__ bf16 Bs[2][64][40];    // [buf][n][k], async-filled, double-buffered

  const int tid  = threadIdx.x;
  const int lane = tid & 31, wave = tid >> 5;
  const int lm   = lane & 15, g = lane >> 4;
  const int m_base = blockIdx.x * 128;
  const int n_base = blockIdx.y * 64;

  // A staging: thread -> row a_mi = tid>>1, 16 consecutive k at a_k0=(tid&1)*16.
  const int a_mi = tid >> 1, a_k0 = (tid & 1) * 16;
  const int a_gm = m_base + a_mi;
  const int a_b  = a_gm / HW;
  const int a_r  = a_gm - a_b * HW;
  const int a_oh = a_r / OW;
  const int a_ow = a_r - a_oh * OW;
  const float* ain = in + (size_t)a_b * Cin * IH * IW;
  const int aih0 = a_oh * stride - pad;
  const int aiw0 = a_ow * stride - pad;

  // B staging: thread -> row b_ni = tid>>2, 8 consecutive k at b_k0=(tid&3)*8.
  const int b_ni = tid >> 2, b_k0 = (tid & 3) * 8;
  const unsigned b_goff = ((unsigned)(n_base + b_ni) * (unsigned)Kp + (unsigned)b_k0) * 2u;
  const unsigned b_lds0 = (unsigned)(size_t)&Bs[0][b_ni][b_k0];  // addr[31:0] = LDS offset
  const unsigned b_lds1 = (unsigned)(size_t)&Bs[1][b_ni][b_k0];

  // preload first B tile into buffer 0 (16B per lane, aligned: Kp mult of 32)
  asm volatile("global_load_async_to_lds_b128 %0, %1, %2 offset:0"
               :: "v"(b_lds0), "v"(b_goff), "s"(wgt) : "memory");

  v8f c[4] = {v8f{}, v8f{}, v8f{}, v8f{}};

  int cur = 0;
  for (int k0 = 0; k0 < Ktot; k0 += 32, cur ^= 1) {
    // ---- stage A (im2col gather, f32 -> bf16), two b128 stores ----
    union { bf16 h[16]; uint4 q[2]; } at;
#pragma unroll
    for (int j = 0; j < 16; ++j) {
      int k = k0 + a_k0 + j;
      float v = 0.f;
      if (k < Ktot) {
        int ci = k / KK;             // constant divisors (KS template param)
        int rr = k - ci * KK;
        int kh = rr / KS;
        int kw = rr - kh * KS;
        int ih = aih0 + kh, iw = aiw0 + kw;
        if (ih >= 0 && ih < IH && iw >= 0 && iw < IW)
          v = ain[((size_t)ci * IH + ih) * IW + iw];
      }
      at.h[j] = (bf16)v;
    }
    *(uint4*)&As[a_mi][a_k0]     = at.q[0];
    *(uint4*)&As[a_mi][a_k0 + 8] = at.q[1];

    // ---- issue async fetch of NEXT B tile, then retire the CURRENT one ----
    if (k0 + 32 < Ktot) {
      unsigned voff = b_goff + (unsigned)(k0 + 32) * 2u;
      unsigned ldst = cur ? b_lds0 : b_lds1;
      asm volatile("global_load_async_to_lds_b128 %0, %1, %2 offset:0"
                   :: "v"(ldst), "v"(voff), "s"(wgt) : "memory");
      asm volatile("s_wait_asynccnt 0x1" ::: "memory");  // in-order: current tile done
    } else {
      asm volatile("s_wait_asynccnt 0x0" ::: "memory");
    }
    __syncthreads();

    // ---- fragments ----
    // A 16-bit 16x32 layout: element e -> K = (e&7) + 16*(e>=8) + 8*g  (contiguous
    //   runs [8g,8g+8) and [16+8g,24+8g)).
    union { v16bf v; uint4 q[2]; } fa;
    fa.q[0] = *(const uint4*)&As[wave * 16 + lm][g * 8];
    fa.q[1] = *(const uint4*)&As[wave * 16 + lm][16 + g * 8];
    // B 16-bit 32x16 layout: element e -> K = (e&7) + 16*(e>=8) + 8*g (same as A).
#pragma unroll
    for (int j = 0; j < 4; ++j) {
      union { v16bf v; uint4 q[2]; } fb;
      fb.q[0] = *(const uint4*)&Bs[cur][j * 16 + lm][g * 8];
      fb.q[1] = *(const uint4*)&Bs[cur][j * 16 + lm][16 + g * 8];
      c[j] = __builtin_amdgcn_wmma_f32_16x16x32_bf16(false, fa.v, false, fb.v,
                                                     (short)0, c[j], false, false);
    }
    __syncthreads();
  }

  // ---- epilogue: D layout M = r + 8*g, N = lane&15; fused BN + residual + relu ----
#pragma unroll
  for (int r = 0; r < 8; ++r) {
    int gm = m_base + wave * 16 + r + g * 8;
    int b  = gm / HW;
    int rr = gm - b * HW;
    int oh = rr / OW;
    int ow = rr - oh * OW;
    size_t obase = (size_t)b * Cout * HW + (size_t)oh * OW + ow;
#pragma unroll
    for (int j = 0; j < 4; ++j) {
      int gn = n_base + j * 16 + lm;
      float v = c[j][r] * scale[gn] + bias[gn];
      size_t oidx = obase + (size_t)gn * HW;
      if (resid) v += resid[oidx];
      if (relu)  v = fmaxf(v, 0.f);
      out[oidx] = v;
    }
  }
}

// ---------------- maxpool 3x3 s2 p1 ----------------
__global__ void k_maxpool(const float* __restrict__ in, float* __restrict__ out,
                          int C, int IH, int IW, int OH, int OW) {
  int i = blockIdx.x * blockDim.x + threadIdx.x;
  int total = BATCH * C * OH * OW;
  if (i >= total) return;
  int ow = i % OW, t = i / OW;
  int oh = t % OH; t /= OH;
  int c  = t % C;  int b = t / C;
  float m = -3.402823466e38f;
  int h0 = oh * 2 - 1, w0 = ow * 2 - 1;
  for (int kh = 0; kh < 3; ++kh) {
    int ih = h0 + kh; if (ih < 0 || ih >= IH) continue;
    for (int kw = 0; kw < 3; ++kw) {
      int iw = w0 + kw; if (iw < 0 || iw >= IW) continue;
      m = fmaxf(m, in[(((size_t)b * C + c) * IH + ih) * IW + iw]);
    }
  }
  out[i] = m;
}

// ---------------- global average pool ----------------
__global__ void k_avgpool(const float* __restrict__ in, float* __restrict__ out,
                          int C, int HW) {
  int i = blockIdx.x * blockDim.x + threadIdx.x;  // b*C + c
  if (i >= BATCH * C) return;
  const float* p = in + (size_t)i * HW;
  float s = 0.f;
  for (int j = 0; j < HW; ++j) s += p[j];
  out[i] = s / (float)HW;
}

// ---------------- FC + gated accumulate into d_out ----------------
__global__ void k_fc(const float* __restrict__ pooled, const float* __restrict__ fw,
                     const float* __restrict__ fb, const float* __restrict__ gates,
                     float* __restrict__ out, int e) {
  int n = threadIdx.x;           // 0..13 (block of 32, guarded)
  int b = blockIdx.x;            // 0..7
  if (n >= NCLS) return;
  float s = fb[n];
  const float* pb = pooled + b * 512;
  for (int c = 0; c < 512; ++c) s += pb[c] * fw[c * NCLS + n];
  out[b * NCLS + n] += gates[b * NEXP + e] * s;
}

// ---------------- orchestration ----------------
extern "C" void kernel_launch(void* const* d_in, const int* in_sizes, int n_in,
                              void* d_out, int out_size, void* d_ws, size_t ws_size,
                              hipStream_t stream) {
  (void)in_sizes; (void)n_in; (void)out_size; (void)ws_size;
  const float* x  = (const float*)d_in[0];
  const float* wg = (const float*)d_in[1];
  float* out = (float*)d_out;
  float* ws  = (float*)d_ws;

  // workspace layout (floats)
  float* gates  = ws;                       // 32
  float* pooled = ws + 32;                  // 8*512
  float* bufS   = ws + 32 + 4096;           // 8*64*192*192 = 18,874,368
  float* bufA   = bufS + 18874368;          // 8*64*96*96 = 4,718,592 each
  float* bufB   = bufA + 4718592;
  float* bufC   = bufB + 4718592;
  float* bufD   = bufC + 4718592;
  bf16*  wbf    = (bf16*)(bufD + 4718592);  // ~11.2M bf16 per expert (reused)

  k_zero<<<1, 128, 0, stream>>>(out, BATCH * NCLS);
  k_gate<<<BATCH, 256, 0, stream>>>(x, wg, gates);

  // pytree leaf index map (sorted dict keys): per block b1,b2,[bd],s1,s2,[sd],w1,w2,[wd]
  struct BI { int b1, b2, bd, s1, s2, sd, w1, w2, wd; };
  const int specs[8][3] = {{64,64,1},{64,64,1},{64,128,2},{128,128,1},
                           {128,256,2},{256,256,1},{256,512,2},{512,512,1}};
  BI bi[8]; int p = 2;
  for (int i = 0; i < 8; ++i) {
    bool dw = (specs[i][2] != 1) || (specs[i][0] != specs[i][1]);
    if (dw) { bi[i] = BI{p, p+1, p+2, p+3, p+4, p+5, p+6, p+7, p+8}; p += 9; }
    else    { bi[i] = BI{p, p+1, -1, p+2, p+3, -1, p+4, p+5, -1};    p += 6; }
  }
  const int i_fcb = p, i_fcw = p + 1, i_stb = p + 2, i_sts = p + 3, i_stw = p + 4;

  // bf16 weight arena offsets (same every expert; region reused per expert)
  size_t wofs = 0;
  auto walloc = [&](size_t nelem) { size_t o = wofs; wofs += nelem; return o; };
  const int stemKp = 160;                   // 147 -> 160 (mult of 32)
  size_t o_stem = walloc((size_t)64 * stemKp);
  struct WO { size_t w1, w2, wd; int kp1, kp2, kpd; };
  WO wo[8];
  for (int i = 0; i < 8; ++i) {
    const int ci = specs[i][0], co = specs[i][1];
    wo[i].kp1 = ci * 9;                     // all multiples of 32 already
    wo[i].kp2 = co * 9;
    wo[i].kpd = ci;
    wo[i].w1 = walloc((size_t)co * wo[i].kp1);
    wo[i].w2 = walloc((size_t)co * wo[i].kp2);
    wo[i].wd = (bi[i].wd >= 0) ? walloc((size_t)co * wo[i].kpd) : 0;
  }

  for (int e = 0; e < NEXP; ++e) {
    // ---- convert this expert's weights to bf16 (zero-padded K rows) ----
    {
      const float* stw = (const float*)d_in[i_stw] + (size_t)e * 64 * 3 * 49;
      int tot = 64 * stemKp;
      k_cvtw<<<(tot + 255) / 256, 256, 0, stream>>>(stw, wbf + o_stem, 147, stemKp, tot);
      for (int i = 0; i < 8; ++i) {
        const int ci = specs[i][0], co = specs[i][1];
        const float* w1 = (const float*)d_in[bi[i].w1] + (size_t)e * co * ci * 9;
        const float* w2 = (const float*)d_in[bi[i].w2] + (size_t)e * co * co * 9;
        int t1n = co * wo[i].kp1;
        int t2n = co * wo[i].kp2;
        k_cvtw<<<(t1n + 255) / 256, 256, 0, stream>>>(w1, wbf + wo[i].w1, ci * 9, wo[i].kp1, t1n);
        k_cvtw<<<(t2n + 255) / 256, 256, 0, stream>>>(w2, wbf + wo[i].w2, co * 9, wo[i].kp2, t2n);
        if (bi[i].wd >= 0) {
          const float* wd = (const float*)d_in[bi[i].wd] + (size_t)e * co * ci;
          int tdn = co * wo[i].kpd;
          k_cvtw<<<(tdn + 255) / 256, 256, 0, stream>>>(wd, wbf + wo[i].wd, ci, wo[i].kpd, tdn);
        }
      }
    }

    // ---- stem: 3x384x384 -> 64x192x192 (7x7 s2 p3) + bn + relu ----
    const float* sts = (const float*)d_in[i_sts] + e * 64;
    const float* stb = (const float*)d_in[i_stb] + e * 64;
    k_conv_wmma<7><<<dim3(BATCH*192*192/128, 1), 256, 0, stream>>>(
        x, wbf + o_stem, sts, stb, nullptr, bufS,
        3, 384, 384, 64, 192, 192, 2, 3, 1, stemKp);
    // maxpool 192 -> 96
    int tot = BATCH * 64 * 96 * 96;
    k_maxpool<<<(tot + 255) / 256, 256, 0, stream>>>(bufS, bufA, 64, 192, 192, 96, 96);

    float* cur = bufA; float* t1 = bufB; float* t2 = bufC; float* t3 = bufD;
    int ih = 96, iw = 96;
    for (int blk = 0; blk < 8; ++blk) {
      const int ci = specs[blk][0], co = specs[blk][1], st = specs[blk][2];
      const int oh = (ih - 1) / st + 1, ow = (iw - 1) / st + 1;   // k3 p1
      const int M = BATCH * oh * ow;
      const float* s1 = (const float*)d_in[bi[blk].s1] + e * co;
      const float* b1 = (const float*)d_in[bi[blk].b1] + e * co;
      const float* s2 = (const float*)d_in[bi[blk].s2] + e * co;
      const float* b2 = (const float*)d_in[bi[blk].b2] + e * co;

      // y1 = relu(bn(conv3x3_s(cur)))
      k_conv_wmma<3><<<dim3(M/128, co/64), 256, 0, stream>>>(
          cur, wbf + wo[blk].w1, s1, b1, nullptr, t1,
          ci, ih, iw, co, oh, ow, st, 1, 1, wo[blk].kp1);
      // shortcut
      const float* res = cur;
      if (bi[blk].wd >= 0) {
        const float* sd = (const float*)d_in[bi[blk].sd] + e * co;
        const float* bd = (const float*)d_in[bi[blk].bd] + e * co;
        k_conv_wmma<1><<<dim3(M/128, co/64), 256, 0, stream>>>(
            cur, wbf + wo[blk].wd, sd, bd, nullptr, t3,
            ci, ih, iw, co, oh, ow, st, 0, 0, wo[blk].kpd);
        res = t3;
      }
      // out = relu(bn(conv3x3(y1)) + shortcut)
      k_conv_wmma<3><<<dim3(M/128, co/64), 256, 0, stream>>>(
          t1, wbf + wo[blk].w2, s2, b2, res, t2,
          co, oh, ow, co, oh, ow, 1, 1, 1, wo[blk].kp2);

      float* old = cur; cur = t2; t2 = old;   // rotate buffers
      ih = oh; iw = ow;
    }
    // global avgpool 12x12 -> (8,512), then fc + gated accumulate
    k_avgpool<<<(BATCH * 512 + 255) / 256, 256, 0, stream>>>(cur, pooled, 512, 144);
    const float* fw = (const float*)d_in[i_fcw] + (size_t)e * 512 * NCLS;
    const float* fb = (const float*)d_in[i_fcb] + e * NCLS;
    k_fc<<<BATCH, 32, 0, stream>>>(pooled, fw, fb, gates, out, e);
  }
}
